// PureCrossAttention_69853348102847
// MI455X (gfx1250) — matrix operand order repaired and checked by reading, not compiled, not run on hardware
//
#include <hip/hip_runtime.h>
#include <hip/hip_bf16.h>
#include <stdint.h>

#define HIDDEN   1024
#define HEADS    16
#define HEAD_DIM 64
#define BSZ      4
#define SEQ      1024
#define MTOT     (BSZ*SEQ)       /* 4096 rows for every GEMM */
#define ATT_SCALE 0.125f         /* 64^-0.5 */

typedef __bf16 bf16_t;
typedef __attribute__((ext_vector_type(16))) __bf16 v16bf;
typedef __attribute__((ext_vector_type(8)))  __bf16 v8bf;
typedef __attribute__((ext_vector_type(8)))  float  v8f;

/* ---- gfx1250 async global->LDS copy path ---- */
#if defined(__has_builtin)
#if __has_builtin(__builtin_amdgcn_global_load_async_to_lds_b128) && \
    __has_builtin(__builtin_amdgcn_s_wait_asynccnt)
#define HAVE_ASYNC_LDS 1
#endif
#endif
#ifndef HAVE_ASYNC_LDS
#define HAVE_ASYNC_LDS 0
#endif

#if HAVE_ASYNC_LDS
typedef int v4i_ld __attribute__((vector_size(16)));
static __device__ __forceinline__ void async_copy16(const void* g, void* l) {
  __builtin_amdgcn_global_load_async_to_lds_b128(
      (__attribute__((address_space(1))) v4i_ld*)(uintptr_t)g,
      (__attribute__((address_space(3))) v4i_ld*)(uint32_t)(uintptr_t)l,
      0, 0);
}
static __device__ __forceinline__ void async_wait0() {
  __builtin_amdgcn_s_wait_asynccnt(0);
}
#endif

static __device__ __forceinline__ unsigned short f2bf(float f) {
  unsigned int u = __float_as_uint(f);
  u += 0x7FFFu + ((u >> 16) & 1u);            // round-to-nearest-even
  return (unsigned short)(u >> 16);
}
static __device__ __forceinline__ v8f v8f_zero() {
  v8f z;
#pragma unroll
  for (int i = 0; i < 8; ++i) z[i] = 0.0f;
  return z;
}
static __device__ __forceinline__ v16bf cat8(v8bf lo, v8bf hi) {
  return __builtin_shufflevector(lo, hi, 0,1,2,3,4,5,6,7,8,9,10,11,12,13,14,15);
}

/* ------------------------------------------------------------------ */
/* fp32 -> bf16 conversion (grid-stride)                               */
/* ------------------------------------------------------------------ */
__global__ void cvt_f32_bf16(const float* __restrict__ in,
                             unsigned short* __restrict__ out, int n) {
  int i = blockIdx.x * blockDim.x + threadIdx.x;
  int stride = gridDim.x * blockDim.x;
  for (; i < n; i += stride) out[i] = f2bf(in[i]);
}

/* ------------------------------------------------------------------ */
/* GEMM: C[M=4096][N=1024] = alpha*(A[M][K=1024] @ W[N][K]^T + bias)   */
/* OUT_MODE 0: fp32 [M][N];  2: bf16 head-split [b][h][s][d]           */
/* Block 256 thr = 8 waves; tile 128(M) x 64(N); K-step 32.            */
/* ------------------------------------------------------------------ */
template<int OUT_MODE>
__global__ __launch_bounds__(256)
void gemm_bf16_wmma(const unsigned short* __restrict__ Au,
                    const unsigned short* __restrict__ Wu,
                    const float* __restrict__ bias,
                    float alpha, void* __restrict__ outp) {
  __shared__ unsigned short AshU[128 * 48];   // A tile, 48-half (96B) padded rows
  __shared__ unsigned short BshU[32 * 72];    // W tile K-major: [k][n], 72-half rows
  const int t    = threadIdx.x;
  const int wave = t >> 5, lane = t & 31;
  const int m0   = blockIdx.y * 128;
  const int n0   = blockIdx.x * 64;

  const uint32_t* Ad = (const uint32_t*)Au;   // K/2 = 512 dwords per row
  const uint32_t* Wd = (const uint32_t*)Wu;
  uint32_t* AshD = (uint32_t*)AshU;
  const bf16_t* AshB = (const bf16_t*)AshU;
  const bf16_t* BshB = (const bf16_t*)BshU;

  v8f acc[4];
#pragma unroll
  for (int i = 0; i < 4; ++i) acc[i] = v8f_zero();

  const int arow = wave * 16 + (lane & 15);
  const int chnk = (lane >> 4) * 8;           // K-chunk base for this lane half

  for (int k0 = 0; k0 < HIDDEN; k0 += 32) {
    /* stage A: 128 rows x 64B (4 x 16B chunks per row) */
#if HAVE_ASYNC_LDS
#pragma unroll
    for (int i = 0; i < 2; ++i) {
      int c = t + 256 * i;                    // 512 chunks
      int row = c >> 2, sub = c & 3;
      async_copy16((const char*)Au + (size_t)(m0 + row) * 2048 + k0 * 2 + sub * 16,
                   (char*)AshU + row * 96 + sub * 16);
    }
#else
#pragma unroll
    for (int i = 0; i < 8; ++i) {
      int idx = t + 256 * i;
      int row = idx >> 4, cd = idx & 15;
      AshD[row * 24 + cd] = Ad[(size_t)(m0 + row) * 512 + (k0 >> 1) + cd];
    }
#endif
    /* stage W transposed into [k][n] (transpose: must go through VGPRs) */
#pragma unroll
    for (int i = 0; i < 4; ++i) {
      int idx = t + 256 * i;
      int nr = idx >> 4, kd = idx & 15;
      uint32_t w = Wd[(size_t)(n0 + nr) * 512 + (k0 >> 1) + kd];
      BshU[(2 * kd + 0) * 72 + nr] = (unsigned short)(w & 0xFFFFu);
      BshU[(2 * kd + 1) * 72 + nr] = (unsigned short)(w >> 16);
    }
    if (k0 + 32 < HIDDEN)                      // gfx1250 global_prefetch_b8
      __builtin_prefetch(&Ad[(size_t)(m0 + (t >> 1)) * 512 + ((k0 + 32) >> 1)], 0, 1);
#if HAVE_ASYNC_LDS
    async_wait0();
#endif
    __syncthreads();

    /* A fragment: lane = M row, two contiguous 8-half chunks */
    v8bf alo = *(const v8bf*)&AshB[arow * 48 + chnk];
    v8bf ahi = *(const v8bf*)&AshB[arow * 48 + 16 + chnk];
    v16bf afrag = cat8(alo, ahi);
#pragma unroll
    for (int nn = 0; nn < 4; ++nn) {
      /* B fragment: lane = N column, K in the same chunk order as A */
      v16bf bfrag;
#pragma unroll
      for (int e = 0; e < 16; ++e) {
        int kk = (e < 8) ? chnk + e : 16 + chnk + (e - 8);
        bfrag[e] = BshB[kk * 72 + nn * 16 + (lane & 15)];
      }
      acc[nn] = __builtin_amdgcn_wmma_f32_16x16x32_bf16(
          false, afrag, false, bfrag, (short)0, acc[nn], false, false);
    }
    __syncthreads();
  }

#pragma unroll
  for (int nn = 0; nn < 4; ++nn) {
#pragma unroll
    for (int r = 0; r < 8; ++r) {
      int m = m0 + wave * 16 + r + 8 * (lane >> 4);
      int n = n0 + nn * 16 + (lane & 15);
      float v = alpha * (acc[nn][r] + bias[n]);
      if (OUT_MODE == 0) {
        ((float*)outp)[(size_t)m * HIDDEN + n] = v;
      } else {
        int b = m >> 10, s = m & 1023;
        int h = n >> 6,  d = n & 63;
        size_t o = (((size_t)(b * HEADS + h)) * SEQ + s) * HEAD_DIM + d;
        ((unsigned short*)outp)[o] = f2bf(v);
      }
    }
  }
}

/* ------------------------------------------------------------------ */
/* Fused attention, 2 passes. Block = 8 waves sharing one (b,h).       */
/* Pass1 keeps lane-local (m,Z) — single cross-lane merge at the end.  */
/* Pass2: O1 += p*V, O2' += exp(-p)*V, z2 += exp(-p); epilogue scales  */
/* O2' by 1/z2. Probabilities never touch HBM.                         */
/* ------------------------------------------------------------------ */
__global__ __launch_bounds__(256)
void attn_fused_wmma(const unsigned short* __restrict__ Qs,
                     const unsigned short* __restrict__ Ks,
                     const unsigned short* __restrict__ Vs,
                     const float* __restrict__ mask,
                     unsigned short* __restrict__ ctx1,
                     unsigned short* __restrict__ ctx2) {
  __shared__ unsigned short KtU[64 * 40];     // K^T tile: [d(64)][kv(32)] pad 40
  __shared__ unsigned short VshU[32 * 72];    // V tile:   [kv(32)][d(64)] pad 72 (144B rows)
  __shared__ unsigned short pb1[8][16 * 32];  // per-wave P transpose bounce
  __shared__ unsigned short pb2[8][16 * 32];
  const int t = threadIdx.x;
  const int wave = t >> 5, lane = t & 31;
  const int bh = blockIdx.x >> 3;             // 64 (b,h) pairs, 8 blocks each
  const int b = bh >> 4, h = bh & 15;
  const int q0 = ((blockIdx.x & 7) * 8 + wave) * 16;

  const bf16_t* Qp = (const bf16_t*)Qs + (((size_t)(b * HEADS + h)) * SEQ + q0) * HEAD_DIM;
  const bf16_t* Kp = (const bf16_t*)Ks + ((size_t)(b * HEADS + h)) * SEQ * HEAD_DIM;
  const bf16_t* Vp = (const bf16_t*)Vs + ((size_t)(b * HEADS + h)) * SEQ * HEAD_DIM;
  const float*  Mp = mask + (size_t)b * SEQ * SEQ;
  const uint32_t* Kd = (const uint32_t*)Kp;   // 32 dwords per kv row
  const uint32_t* Vd = (const uint32_t*)Vp;
  const bf16_t* KtB = (const bf16_t*)KtU;
  const bf16_t* VshB = (const bf16_t*)VshU;
  uint32_t* VshD = (uint32_t*)VshU;

  const int qrow = lane & 15;
  const int chnk = (lane >> 4) * 8;

  /* Q A-fragments (Q pre-scaled by 1/8 in projection) */
  v16bf aq[2];
#pragma unroll
  for (int s = 0; s < 2; ++s) {
    v8bf lo = *(const v8bf*)&Qp[qrow * 64 + 32 * s + chnk];
    v8bf hi = *(const v8bf*)&Qp[qrow * 64 + 32 * s + 16 + chnk];
    aq[s] = cat8(lo, hi);
  }

  auto stageKt = [&](int kv0) {               // transpose to [d][kv]
#pragma unroll
    for (int i = 0; i < 4; ++i) {
      int idx = t + 256 * i;                  // 1024 dwords
      int kvr = idx >> 5, cd = idx & 31;
      uint32_t w = Kd[(size_t)(kv0 + kvr) * 32 + cd];
      KtU[(2 * cd + 0) * 40 + kvr] = (unsigned short)(w & 0xFFFFu);
      KtU[(2 * cd + 1) * 40 + kvr] = (unsigned short)(w >> 16);
    }
  };
  auto stageV = [&](int kv0) {                // row-major copy (async if available)
#if HAVE_ASYNC_LDS
    int kvr = t >> 3, sub = t & 7;            // 256 chunks of 16B
    async_copy16((const char*)Vp + (size_t)(kv0 + kvr) * 128 + sub * 16,
                 (char*)VshU + kvr * 144 + sub * 16);
#else
#pragma unroll
    for (int i = 0; i < 4; ++i) {
      int idx = t + 256 * i;
      int kvr = idx >> 5, cd = idx & 31;
      VshD[kvr * 36 + cd] = Vd[(size_t)(kv0 + kvr) * 32 + cd];
    }
#endif
  };

  /* S tile for 16-kv half tt of the staged 32; mask added from global */
  auto computeS = [&](int tt, int kv0abs) -> v8f {
    v8f S = v8f_zero();
#pragma unroll
    for (int s = 0; s < 2; ++s) {
      v16bf bk;
#pragma unroll
      for (int e = 0; e < 16; ++e) {
        int kk = (e < 8) ? chnk + e : 16 + chnk + (e - 8);
        bk[e] = KtB[(32 * s + kk) * 40 + 16 * tt + (lane & 15)];
      }
      S = __builtin_amdgcn_wmma_f32_16x16x32_bf16(
          false, aq[s], false, bk, (short)0, S, false, false);
    }
#pragma unroll
    for (int r = 0; r < 8; ++r)
      S[r] += Mp[(size_t)(q0 + r + 8 * (lane >> 4)) * SEQ + kv0abs + (lane & 15)];
    return S;
  };
  auto rsum16 = [&](float v) {
#pragma unroll
    for (int mm = 8; mm; mm >>= 1) v += __shfl_xor(v, mm, 16);
    return v;
  };

  /* ---- pass 1: lane-local online (m,Z); one cross-lane merge ---- */
  float m[8], Z[8];
#pragma unroll
  for (int r = 0; r < 8; ++r) { m[r] = -1e30f; Z[r] = 0.0f; }
  for (int kv0 = 0; kv0 < SEQ; kv0 += 32) {
    __syncthreads();
    stageKt(kv0);
    __syncthreads();
#pragma unroll
    for (int tt = 0; tt < 2; ++tt) {
      v8f S = computeS(tt, kv0 + 16 * tt);
#pragma unroll
      for (int r = 0; r < 8; ++r) {
        float mn = fmaxf(m[r], S[r]);
        Z[r] = Z[r] * __expf(m[r] - mn) + __expf(S[r] - mn);
        m[r] = mn;
      }
    }
  }
  /* merge (m,Z) pairs across the 16 lanes of each half */
#pragma unroll
  for (int r = 0; r < 8; ++r) {
#pragma unroll
    for (int mm = 8; mm; mm >>= 1) {
      float mo = __shfl_xor(m[r], mm, 16);
      float Zo = __shfl_xor(Z[r], mm, 16);
      float mn = fmaxf(m[r], mo);
      Z[r] = Z[r] * __expf(m[r] - mn) + Zo * __expf(mo - mn);
      m[r] = mn;
    }
  }
  float invZ[8];
#pragma unroll
  for (int r = 0; r < 8; ++r) invZ[r] = 1.0f / Z[r];

  /* ---- pass 2: O1 = P@V, O2' = exp(-P)@V, z2 = row-sum exp(-P) ---- */
  v8f O1[4], O2[4];
#pragma unroll
  for (int i = 0; i < 4; ++i) { O1[i] = v8f_zero(); O2[i] = v8f_zero(); }
  float z2p[8];
#pragma unroll
  for (int r = 0; r < 8; ++r) z2p[r] = 0.0f;

  for (int kv0 = 0; kv0 < SEQ; kv0 += 32) {
    __syncthreads();
    stageKt(kv0);
    stageV(kv0);
#if HAVE_ASYNC_LDS
    async_wait0();
#endif
    __syncthreads();
#pragma unroll
    for (int tt = 0; tt < 2; ++tt) {
      v8f S = computeS(tt, kv0 + 16 * tt);
#pragma unroll
      for (int r = 0; r < 8; ++r) {
        float p  = __expf(S[r] - m[r]) * invZ[r];
        float w2 = __expf(-p);
        z2p[r] += w2;
        int ql  = r + 8 * (lane >> 4);
        int col = 16 * tt + (lane & 15);
        pb1[wave][ql * 32 + col] = f2bf(p);   // C-layout -> row-major bounce
        pb2[wave][ql * 32 + col] = f2bf(w2);
      }
    }
    /* rebuild as A fragments — same-wave DS ops stay in order */
    const bf16_t* p1 = (const bf16_t*)pb1[wave];
    const bf16_t* p2 = (const bf16_t*)pb2[wave];
    v16bf aP1 = cat8(*(const v8bf*)&p1[qrow * 32 + chnk],
                     *(const v8bf*)&p1[qrow * 32 + 16 + chnk]);
    v16bf aP2 = cat8(*(const v8bf*)&p2[qrow * 32 + chnk],
                     *(const v8bf*)&p2[qrow * 32 + 16 + chnk]);
#pragma unroll
    for (int nn = 0; nn < 4; ++nn) {
      v16bf bV;                                         // lane = d column
#pragma unroll
      for (int e = 0; e < 16; ++e) {
        int kk = (e < 8) ? chnk + e : 16 + chnk + (e - 8);
        bV[e] = VshB[kk * 72 + nn * 16 + (lane & 15)];
      }
      O1[nn] = __builtin_amdgcn_wmma_f32_16x16x32_bf16(
          false, aP1, false, bV, (short)0, O1[nn], false, false);
      O2[nn] = __builtin_amdgcn_wmma_f32_16x16x32_bf16(
          false, aP2, false, bV, (short)0, O2[nn], false, false);
    }
  }

  float invZ2[8];
#pragma unroll
  for (int r = 0; r < 8; ++r) invZ2[r] = 1.0f / rsum16(z2p[r]);

  /* write merged-head bf16 context [b][q][h*64+d]; O2 scaled per row */
#pragma unroll
  for (int nn = 0; nn < 4; ++nn) {
#pragma unroll
    for (int r = 0; r < 8; ++r) {
      int qq  = q0 + r + 8 * (lane >> 4);
      int col = h * 64 + nn * 16 + (lane & 15);
      size_t o = ((size_t)b * SEQ + qq) * HIDDEN + col;
      ctx1[o] = f2bf(O1[nn][r]);
      ctx2[o] = f2bf(O2[nn][r] * invZ2[r]);
    }
  }
}

/* ------------------------------------------------------------------ */
extern "C" void kernel_launch(void* const* d_in, const int* in_sizes, int n_in,
                              void* d_out, int out_size, void* d_ws, size_t ws_size,
                              hipStream_t stream) {
  const float* q   = (const float*)d_in[0];
  const float* kv  = (const float*)d_in[1];
  const float* msk = (const float*)d_in[2];
  const float* Wq  = (const float*)d_in[3];
  const float* bq  = (const float*)d_in[4];
  const float* Wk  = (const float*)d_in[5];
  const float* bk  = (const float*)d_in[6];
  const float* Wv  = (const float*)d_in[7];
  const float* bv  = (const float*)d_in[8];
  const float* Wo  = (const float*)d_in[9];
  const float* bo  = (const float*)d_in[10];
  const float* Wr  = (const float*)d_in[11];
  const float* br  = (const float*)d_in[12];
  float* out = (float*)d_out;

  const size_t NA = (size_t)MTOT * HIDDEN;    // 4M activation elems
  const size_t NW = (size_t)HIDDEN * HIDDEN;  // 1M weight elems
  char* w = (char*)d_ws;
  unsigned short* qbf  = (unsigned short*)w; w += NA * 2;
  unsigned short* kvbf = (unsigned short*)w; w += NA * 2;
  unsigned short* wqb  = (unsigned short*)w; w += NW * 2;
  unsigned short* wkb  = (unsigned short*)w; w += NW * 2;
  unsigned short* wvb  = (unsigned short*)w; w += NW * 2;
  unsigned short* wob  = (unsigned short*)w; w += NW * 2;
  unsigned short* wrb  = (unsigned short*)w; w += NW * 2;
  unsigned short* Qsb  = (unsigned short*)w; w += NA * 2;
  unsigned short* Ksb  = (unsigned short*)w; w += NA * 2;
  unsigned short* Vsb  = (unsigned short*)w; w += NA * 2;
  unsigned short* ctx1 = (unsigned short*)w; w += NA * 2;
  unsigned short* ctx2 = (unsigned short*)w; w += NA * 2;

  dim3 cb(256), cg(512);
  cvt_f32_bf16<<<cg, cb, 0, stream>>>(q,  qbf,  (int)NA);
  cvt_f32_bf16<<<cg, cb, 0, stream>>>(kv, kvbf, (int)NA);
  cvt_f32_bf16<<<cg, cb, 0, stream>>>(Wq, wqb, (int)NW);
  cvt_f32_bf16<<<cg, cb, 0, stream>>>(Wk, wkb, (int)NW);
  cvt_f32_bf16<<<cg, cb, 0, stream>>>(Wv, wvb, (int)NW);
  cvt_f32_bf16<<<cg, cb, 0, stream>>>(Wo, wob, (int)NW);
  cvt_f32_bf16<<<cg, cb, 0, stream>>>(Wr, wrb, (int)NW);

  dim3 gb(256), gg(HIDDEN / 64, MTOT / 128);  // (16, 32)
  gemm_bf16_wmma<2><<<gg, gb, 0, stream>>>(qbf,  wqb, bq, ATT_SCALE, Qsb);
  gemm_bf16_wmma<2><<<gg, gb, 0, stream>>>(kvbf, wkb, bk, 1.0f,      Ksb);
  gemm_bf16_wmma<2><<<gg, gb, 0, stream>>>(kvbf, wvb, bv, 1.0f,      Vsb);

  attn_fused_wmma<<<dim3(512), dim3(256), 0, stream>>>(Qsb, Ksb, Vsb, msk, ctx1, ctx2);

  gemm_bf16_wmma<0><<<gg, gb, 0, stream>>>(ctx1, wob, bo, 1.0f, (void*)out);
  gemm_bf16_wmma<0><<<gg, gb, 0, stream>>>(ctx2, wrb, br, 1.0f, (void*)(out + NA));
}
